// TabM_Layer_87531433492987
// MI455X (gfx1250) — matrix operand order repaired and checked by reading, not compiled, not run on hardware
//
#include <hip/hip_runtime.h>

typedef __attribute__((ext_vector_type(16))) _Float16 v16h;
typedef __attribute__((ext_vector_type(8)))  float    v8f;
typedef __attribute__((ext_vector_type(4)))  _Float16 h4;

#define BDIM  4096
#define D_IN  1024
#define D_OUT 1024
#define KENS  8

// ---------------- stage 1a: xh = f16(x) ----------------
__global__ void cvt_x_kernel(const float* __restrict__ x, _Float16* __restrict__ xh) {
  const int i = (blockIdx.x * blockDim.x + threadIdx.x) * 4;
  const float4 v = *(const float4*)(x + i);
  h4 o = { (_Float16)v.x, (_Float16)v.y, (_Float16)v.z, (_Float16)v.w };
  *(h4*)(xh + i) = o;
}

// ---------------- stage 1b: Wk[k][o][i] = f16(W[o][i] * r[k][i]) ----------------
__global__ void build_wk_kernel(const float* __restrict__ W, const float* __restrict__ r,
                                _Float16* __restrict__ wk) {
  const int k   = blockIdx.z;
  const int idx = blockIdx.x * blockDim.x + threadIdx.x;   // 0 .. 1024*256-1
  const int o   = idx >> 8;
  const int ic  = (idx & 255) * 4;
  const float4 w  = *(const float4*)(W + o * D_IN + ic);
  const float4 rr = *(const float4*)(r + k * D_IN + ic);
  h4 out = { (_Float16)(w.x * rr.x), (_Float16)(w.y * rr.y),
             (_Float16)(w.z * rr.z), (_Float16)(w.w * rr.w) };
  *(h4*)(wk + ((size_t)k << 20) + (size_t)o * D_IN + ic) = out;
}

// ---------------- stage 2: per-k GEMM with WMMA f16 + async global->LDS ----------------
// Block tile 256(M) x 128(N), 8 wave32 each computing 64x64 (4x4 WMMA tiles).
#define TK       32
#define LDS_ROW  80                    // 64 data bytes + 16 pad -> conflict-free b128 frag reads
#define LDS_A    (256 * LDS_ROW)       // 20480 B (A tile 256x32 f16)
#define LDS_B    (128 * LDS_ROW)       // 10240 B (B tile 128x32 f16)
#define LDS_BUF  (LDS_A + LDS_B)       // 30720 B per buffer

// async copy of 16 bytes global -> LDS, tracked by ASYNCcnt (no VGPR data)
__device__ __forceinline__ void async_g2l_b128(unsigned lds_addr, const void* gaddr) {
  asm volatile("global_load_async_to_lds_b128 %0, %1, off"
               :: "v"(lds_addr), "v"((unsigned long long)gaddr)
               : "memory");
}
__device__ __forceinline__ void wait_async0() {
  asm volatile("s_wait_asynccnt 0x0" ::: "memory");
}

__launch_bounds__(256)
__global__ void tabm_gemm_kernel(const _Float16* __restrict__ A,   // [4096][1024] f16
                                 const _Float16* __restrict__ Wk,  // [8][1024][1024] f16
                                 const float* __restrict__ bias,   // [1024]
                                 const float* __restrict__ s,      // [8][1024]
                                 float* __restrict__ out) {        // [4096*8][1024]
  __shared__ __align__(16) unsigned char lds[2 * LDS_BUF];         // 60 KB

  // 32-bit LDS-space base address of the shared array
  typedef __attribute__((address_space(3))) unsigned char lds_uc;
  const unsigned ldsBase = (unsigned)(unsigned long long)(lds_uc*)lds;

  const int tid   = threadIdx.x;
  const int lane  = tid & 31;
  const int waveM = (tid >> 5) & 3;   // 0..3 -> 64-row slab
  const int waveN = tid >> 7;         // 0..1 -> 64-col slab

  const int bM = blockIdx.x * 256;    // batch-row tile
  const int bN = blockIdx.y * 128;    // output-col tile
  const int kz = blockIdx.z;          // ensemble member

  const _Float16* Abase = A  + (size_t)bM * D_IN;
  const _Float16* Bbase = Wk + ((size_t)kz << 20) + (size_t)bN * D_IN;

  // tile loaders: A = 1024 16B chunks (4/thread), B = 512 chunks (2/thread)
  const int lrow = tid >> 2;          // 0..63
  const int cch  = tid & 3;           // 16B chunk within 64B row
  unsigned ldsA[4], ldsB[2];
#pragma unroll
  for (int j = 0; j < 4; ++j)
    ldsA[j] = ldsBase + (unsigned)((lrow + 64 * j) * LDS_ROW + cch * 16);
#pragma unroll
  for (int j = 0; j < 2; ++j)
    ldsB[j] = ldsBase + (unsigned)(LDS_A + (lrow + 64 * j) * LDS_ROW + cch * 16);
  const _Float16* ga = Abase + (size_t)lrow * D_IN + cch * 8;
  const _Float16* gb = Bbase + (size_t)lrow * D_IN + cch * 8;

  // WMMA fragment LDS addressing (ISA 7.12.2 layouts)
  const int frow    = lane & 15;
  const int halfsel = lane >> 4;
  // A 16x32 f16: lane holds K[0..7]&K[16..23] (half 0) or K[8..15]&K[24..31] (half 1)
  const unsigned aoff = (unsigned)((waveM * 64 + frow) * LDS_ROW + halfsel * 16);
  // B 32x16 f16: lane holds K[0..7]&K[16..23] (half 0) or K[8..15]&K[24..31] (half 1)
  const unsigned boff = (unsigned)(LDS_A + (waveN * 64 + frow) * LDS_ROW + halfsel * 16);

  v8f acc[4][4];
#pragma unroll
  for (int mt = 0; mt < 4; ++mt)
#pragma unroll
    for (int nt = 0; nt < 4; ++nt) acc[mt][nt] = {};

  // prologue: async-fill buffer 0 (K-slab 0)
#pragma unroll
  for (int j = 0; j < 4; ++j) async_g2l_b128(ldsA[j], ga + (size_t)(64 * j) * D_IN);
#pragma unroll
  for (int j = 0; j < 2; ++j) async_g2l_b128(ldsB[j], gb + (size_t)(64 * j) * D_IN);
  wait_async0();
  __syncthreads();

  unsigned curOff = 0, nxtOff = LDS_BUF;
  const int KT = D_IN / TK;   // 32 K-steps
#pragma unroll 1
  for (int kt = 0; kt < KT; ++kt) {
    if (kt + 1 < KT) {        // async-fill next buffer while computing (no VGPRs held)
      ga += TK;
      gb += TK;
#pragma unroll
      for (int j = 0; j < 4; ++j) async_g2l_b128(ldsA[j] + nxtOff, ga + (size_t)(64 * j) * D_IN);
#pragma unroll
      for (int j = 0; j < 2; ++j) async_g2l_b128(ldsB[j] + nxtOff, gb + (size_t)(64 * j) * D_IN);
    }

    const unsigned char* la = lds + curOff;
    union FragU { v16h h; int4 q[2]; };
    FragU fa[4], fb[4];
#pragma unroll
    for (int mt = 0; mt < 4; ++mt) {
      fa[mt].q[0] = *(const int4*)(la + aoff + mt * 16 * LDS_ROW);
      fa[mt].q[1] = *(const int4*)(la + aoff + mt * 16 * LDS_ROW + 32);
    }
#pragma unroll
    for (int nt = 0; nt < 4; ++nt) {
      fb[nt].q[0] = *(const int4*)(la + boff + nt * 16 * LDS_ROW);
      fb[nt].q[1] = *(const int4*)(la + boff + nt * 16 * LDS_ROW + 32);
    }
#pragma unroll
    for (int nt = 0; nt < 4; ++nt)
#pragma unroll
      for (int mt = 0; mt < 4; ++mt)
        acc[mt][nt] = __builtin_amdgcn_wmma_f32_16x16x32_f16(
            false, fa[mt].h, false, fb[nt].h, (short)0, acc[mt][nt], false, false);

    if (kt + 1 < KT) {
      wait_async0();          // our async fills of nxt are done
      __syncthreads();        // everyone's fills are done / everyone done reading cur
    }
    const unsigned t = curOff; curOff = nxtOff; nxtOff = t;
  }

  // epilogue: out[(m*8+kz)*1024 + o] = acc*s + bias*s
#pragma unroll
  for (int nt = 0; nt < 4; ++nt) {
    const int o    = bN + waveN * 64 + nt * 16 + frow;
    const float so = s[kz * D_OUT + o];
    const float bs = bias[o] * so;
#pragma unroll
    for (int mt = 0; mt < 4; ++mt) {
      const int mbase = bM + waveM * 64 + mt * 16 + halfsel * 8;
#pragma unroll
      for (int v = 0; v < 8; ++v) {
        const int m = mbase + v;
        out[((size_t)m * KENS + kz) * D_OUT + o] = acc[mt][nt][v] * so + bs;
      }
    }
  }
}

extern "C" void kernel_launch(void* const* d_in, const int* in_sizes, int n_in,
                              void* d_out, int out_size, void* d_ws, size_t ws_size,
                              hipStream_t stream) {
  const float* x    = (const float*)d_in[0];   // [4096,1024]
  const float* W    = (const float*)d_in[1];   // [1024,1024]
  const float* bias = (const float*)d_in[2];   // [1024]
  const float* r    = (const float*)d_in[3];   // [8,1024]
  const float* s    = (const float*)d_in[4];   // [8,1024]
  float* out = (float*)d_out;

  _Float16* xh = (_Float16*)d_ws;                                                   // 8 MB
  _Float16* wk = (_Float16*)((char*)d_ws + (size_t)BDIM * D_IN * sizeof(_Float16)); // 16 MB

  cvt_x_kernel<<<dim3(BDIM * D_IN / 1024), 256, 0, stream>>>(x, xh);
  build_wk_kernel<<<dim3(D_OUT * D_IN / 1024, 1, KENS), 256, 0, stream>>>(W, r, wk);
  tabm_gemm_kernel<<<dim3(BDIM / 256, D_OUT / 128, KENS), 256, 0, stream>>>(xh, wk, bias, s, out);
}
